// Method_GCN_Citeseer_44418551775395
// MI455X (gfx1250) — matrix operand order repaired and checked by reading, not compiled, not run on hardware
//
#include <hip/hip_runtime.h>
#include <hip/hip_bf16.h>
#include <stdint.h>
#include <stddef.h>

// ---------------------------------------------------------------------------
// GCN (4-layer) for MI455X / gfx1250, wave32.
//   support = X @ W            -> v_wmma_f32_16x16x32_bf16 (f32 accumulate),
//                                 A tile staged via global_load_async_to_lds
//   agg[r] += w_e * support[c] -> L2-resident gather + global_atomic_add_f32
//   h = relu(agg + b)          -> fused bias/relu/bf16-convert
// ---------------------------------------------------------------------------

typedef __bf16 bf16;
typedef bf16  v16bf __attribute__((ext_vector_type(16)));
typedef bf16  v8bf  __attribute__((ext_vector_type(8)));
typedef float v8f   __attribute__((ext_vector_type(8)));

#define N_NODES 100000
#define N_EDGES 1600000
#define F_IN    512
#define HID     128
#define N_CLASS 6

// ---- weight transpose + convert: Wt[o*K + k] = bf16(W[k*Nout + o]) ---------
// Rows o >= Nout are zero-padded (pads layer-4's 6 cols to a full 16 tile).
__global__ void k_cvt_w_t(const float* __restrict__ W, bf16* __restrict__ Wt,
                          int K, int Nout, int NoutPad) {
  int i = blockIdx.x * blockDim.x + threadIdx.x;
  if (i >= NoutPad * K) return;
  int o = i / K, k = i - o * K;
  float v = (o < Nout) ? W[k * Nout + o] : 0.0f;
  Wt[i] = (bf16)v;
}

__global__ void k_zero(float* __restrict__ p, int n) {
  int i = blockIdx.x * blockDim.x + threadIdx.x;
  if (i < n) p[i] = 0.0f;
}

// ---- fp32 fragment load + in-register bf16 convert (layer 1 only) ----------
__device__ __forceinline__ v8bf load8_f32(const float* p) {
  float4 a = *(const float4*)p;
  float4 b = *(const float4*)(p + 4);
  v8bf r;
  r[0]=(bf16)a.x; r[1]=(bf16)a.y; r[2]=(bf16)a.z; r[3]=(bf16)a.w;
  r[4]=(bf16)b.x; r[5]=(bf16)b.y; r[6]=(bf16)b.z; r[7]=(bf16)b.w;
  return r;
}

__device__ __forceinline__ v8f wmma_bf16(v8bf alo, v8bf ahi, v8bf blo, v8bf bhi, v8f acc) {
  v16bf a = __builtin_shufflevector(alo, ahi, 0,1,2,3,4,5,6,7,8,9,10,11,12,13,14,15);
  v16bf b = __builtin_shufflevector(blo, bhi, 0,1,2,3,4,5,6,7,8,9,10,11,12,13,14,15);
  return __builtin_amdgcn_wmma_f32_16x16x32_bf16(
      /*neg_a=*/false, a, /*neg_b=*/false, b,
      /*c_mod=*/(short)0, acc, /*reuse_a=*/false, /*reuse_b=*/false);
}

// ---- WMMA GEMM, layer 1: A = fp32 x[M x 512], converted in-register --------
// grid.x = M/16 row tiles; blockDim.x = (Nout/16)*32; wave w owns col-tile w.
__global__ void k_gemm_wmma_f32(const float* __restrict__ A, const bf16* __restrict__ Wt,
                                float* __restrict__ C, int K, int Nout) {
  const int lane  = threadIdx.x & 31;
  const int ntile = threadIdx.x >> 5;
  const int mtile = blockIdx.x;
  const int mrow  = (mtile << 4) + (lane & 15);
  const int kA    = (lane >> 4) << 3;               // A: lanes 16-31 -> K+8
  const int kB    = (lane >> 4) << 3;               // B: lanes 16-31 -> K+8
  const int ncol  = (ntile << 4) + (lane & 15);
  const float* a0 = A  + (size_t)mrow * K + kA;
  const bf16*  b0 = Wt + (size_t)ncol * K + kB;

  v8f acc = {};
  for (int k0 = 0; k0 < K; k0 += 32) {
    v8bf alo = load8_f32(a0 + k0);
    v8bf ahi = load8_f32(a0 + k0 + 16);
    v8bf blo = *(const v8bf*)(b0 + k0);
    v8bf bhi = *(const v8bf*)(b0 + k0 + 16);
    acc = wmma_bf16(alo, ahi, blo, bhi, acc);
  }
  const int mbase = (mtile << 4) + ((lane >> 4) << 3);
#pragma unroll
  for (int i = 0; i < 8; ++i)
    C[(size_t)(mbase + i) * Nout + ncol] = acc[i];
}

// ---- WMMA GEMM, layers 2-4: A = bf16 h[M x 128], tile staged in LDS --------
// The 16x128 bf16 A tile is one contiguous 4KB block -> async copy to LDS
// (ASYNCcnt), then every wave reads its fragments via ds_load_b128 instead of
// 8x-redundant VMEM traffic. Wave-uniform control flow -> EXEC all-ones for
// every WMMA.
__global__ void k_gemm_wmma_lds(const bf16* __restrict__ A, const bf16* __restrict__ Wt,
                                float* __restrict__ C, int Nout) {
  __shared__ __align__(16) bf16 tileA[16 * HID];    // 4 KB

  const int mtile = blockIdx.x;
  const bf16* gsrc = A + (size_t)mtile * 16 * HID;  // contiguous 4096 B

  // stage: (16*128*2)/16 = 256 b128 chunks spread over the block
  for (int i = threadIdx.x; i < (16 * HID) / 8; i += blockDim.x) {
    uint32_t lds_addr = (uint32_t)(uintptr_t)&tileA[i * 8];  // low 32b = LDS offset
    const bf16* g = gsrc + i * 8;
    asm volatile("global_load_async_to_lds_b128 %0, %1, off"
                 :: "v"(lds_addr), "v"(g) : "memory");
  }
  asm volatile("s_wait_asynccnt 0" ::: "memory");
  __syncthreads();

  const int lane  = threadIdx.x & 31;
  const int ntile = threadIdx.x >> 5;
  const int kA    = (lane >> 4) << 3;
  const int kB    = (lane >> 4) << 3;
  const int ncol  = (ntile << 4) + (lane & 15);
  const bf16* a0  = &tileA[(lane & 15) * HID + kA]; // LDS
  const bf16* b0  = Wt + (size_t)ncol * HID + kB;   // global (WGP$/L2 resident)

  v8f acc = {};
#pragma unroll
  for (int k0 = 0; k0 < HID; k0 += 32) {
    v8bf alo = *(const v8bf*)(a0 + k0);             // ds_load_b128
    v8bf ahi = *(const v8bf*)(a0 + k0 + 16);
    v8bf blo = *(const v8bf*)(b0 + k0);
    v8bf bhi = *(const v8bf*)(b0 + k0 + 16);
    acc = wmma_bf16(alo, ahi, blo, bhi, acc);
  }
  const int mbase = (mtile << 4) + ((lane >> 4) << 3);
#pragma unroll
  for (int i = 0; i < 8; ++i)
    C[(size_t)(mbase + i) * Nout + ncol] = acc[i];
}

// ---- SpMM (F=128): one wave per edge, each lane handles 4 features ---------
__global__ void k_spmm128(const int* __restrict__ row, const int* __restrict__ col,
                          const float* __restrict__ w, const float* __restrict__ sup,
                          float* __restrict__ agg) {
  int e    = (blockIdx.x * blockDim.x + threadIdx.x) >> 5;
  int lane = threadIdx.x & 31;
  if (e >= N_EDGES) return;
  int r = row[e], c = col[e];
  float we = w[e];
  float4 v = ((const float4*)(sup + (size_t)c * HID))[lane];   // L2-resident
  float* o = agg + (size_t)r * HID + lane * 4;
  atomicAdd(o + 0, v.x * we);
  atomicAdd(o + 1, v.y * we);
  atomicAdd(o + 2, v.z * we);
  atomicAdd(o + 3, v.w * we);
}

// ---- SpMM (F=6, sup stride 16, agg stride 8) -------------------------------
__global__ void k_spmm6(const int* __restrict__ row, const int* __restrict__ col,
                        const float* __restrict__ w, const float* __restrict__ sup,
                        float* __restrict__ agg) {
  int idx = blockIdx.x * blockDim.x + threadIdx.x;
  int e = idx >> 3, f = idx & 7;
  if (e >= N_EDGES || f >= N_CLASS) return;
  int r = row[e], c = col[e];
  atomicAdd(agg + (size_t)r * 8 + f, w[e] * sup[(size_t)c * 16 + f]);
}

// ---- fused bias + relu + convert-to-bf16 (next layer's A) ------------------
__global__ void k_bias_relu_bf16(const float* __restrict__ agg, const float* __restrict__ b,
                                 bf16* __restrict__ h) {
  int i = blockIdx.x * blockDim.x + threadIdx.x;
  if (i >= N_NODES * HID) return;
  float v = agg[i] + b[i & (HID - 1)];
  h[i] = (bf16)(v > 0.0f ? v : 0.0f);
}

// ---- final bias (no relu), agg stride 8 -> tightly packed output -----------
__global__ void k_final(const float* __restrict__ agg, const float* __restrict__ b4,
                        float* __restrict__ out) {
  int i = blockIdx.x * blockDim.x + threadIdx.x;
  if (i >= N_NODES * N_CLASS) return;
  int n = i / N_CLASS, f = i - n * N_CLASS;
  out[i] = agg[(size_t)n * 8 + f] + b4[f];
}

// ---------------------------------------------------------------------------
extern "C" void kernel_launch(void* const* d_in, const int* in_sizes, int n_in,
                              void* d_out, int out_size, void* d_ws, size_t ws_size,
                              hipStream_t stream) {
  (void)in_sizes; (void)n_in; (void)out_size; (void)ws_size;

  const float* x    = (const float*)d_in[0];
  const int*   erow = (const int*)  d_in[1];
  const int*   ecol = (const int*)  d_in[2];
  const float* ew   = (const float*)d_in[3];
  const float* W1 = (const float*)d_in[4];  const float* b1 = (const float*)d_in[5];
  const float* W2 = (const float*)d_in[6];  const float* b2 = (const float*)d_in[7];
  const float* W3 = (const float*)d_in[8];  const float* b3 = (const float*)d_in[9];
  const float* W4 = (const float*)d_in[10]; const float* b4 = (const float*)d_in[11];
  float* out = (float*)d_out;

  // ---- workspace carve (256B aligned); everything read is rewritten each call
  char* ws = (char*)d_ws;
  size_t off = 0;
  auto carve = [&](size_t bytes) -> char* {
    char* p = ws + off;
    off += (bytes + 255) & ~(size_t)255;
    return p;
  };
  bf16*  wt1  = (bf16*) carve((size_t)HID * F_IN * sizeof(bf16));   // [128][512]
  bf16*  wt2  = (bf16*) carve((size_t)HID * HID  * sizeof(bf16));   // [128][128]
  bf16*  wt3  = (bf16*) carve((size_t)HID * HID  * sizeof(bf16));   // [128][128]
  bf16*  wt4  = (bf16*) carve((size_t)16  * HID  * sizeof(bf16));   // [16][128] pad
  float* sup  = (float*)carve((size_t)N_NODES * HID * sizeof(float));
  float* agg  = (float*)carve((size_t)N_NODES * HID * sizeof(float));
  bf16*  h    = (bf16*) carve((size_t)N_NODES * HID * sizeof(bf16));
  float* sup4 = (float*)carve((size_t)N_NODES * 16 * sizeof(float));
  float* agg4 = (float*)carve((size_t)N_NODES * 8  * sizeof(float));

  const int MT = N_NODES / 16;  // 6250 row tiles (exact)

  // ---- per-call weight convert/transpose (cheap: <0.2 MB total)
  k_cvt_w_t<<<(HID * F_IN + 255) / 256, 256, 0, stream>>>(W1, wt1, F_IN, HID, HID);
  k_cvt_w_t<<<(HID * HID  + 255) / 256, 256, 0, stream>>>(W2, wt2, HID, HID, HID);
  k_cvt_w_t<<<(HID * HID  + 255) / 256, 256, 0, stream>>>(W3, wt3, HID, HID, HID);
  k_cvt_w_t<<<(16  * HID  + 255) / 256, 256, 0, stream>>>(W4, wt4, HID, N_CLASS, 16);

  // ---- layer 1 (A = fp32 x, converted to bf16 in-register; x read once)
  k_gemm_wmma_f32<<<MT, 256, 0, stream>>>(x, wt1, sup, F_IN, HID);
  k_zero<<<(N_NODES * HID) / 256, 256, 0, stream>>>(agg, N_NODES * HID);
  k_spmm128<<<(N_EDGES * 32) / 256, 256, 0, stream>>>(erow, ecol, ew, sup, agg);
  k_bias_relu_bf16<<<(N_NODES * HID) / 256, 256, 0, stream>>>(agg, b1, h);

  // ---- layer 2 (A tile async-staged to LDS)
  k_gemm_wmma_lds<<<MT, 256, 0, stream>>>(h, wt2, sup, HID);
  k_zero<<<(N_NODES * HID) / 256, 256, 0, stream>>>(agg, N_NODES * HID);
  k_spmm128<<<(N_EDGES * 32) / 256, 256, 0, stream>>>(erow, ecol, ew, sup, agg);
  k_bias_relu_bf16<<<(N_NODES * HID) / 256, 256, 0, stream>>>(agg, b2, h);

  // ---- layer 3
  k_gemm_wmma_lds<<<MT, 256, 0, stream>>>(h, wt3, sup, HID);
  k_zero<<<(N_NODES * HID) / 256, 256, 0, stream>>>(agg, N_NODES * HID);
  k_spmm128<<<(N_EDGES * 32) / 256, 256, 0, stream>>>(erow, ecol, ew, sup, agg);
  k_bias_relu_bf16<<<(N_NODES * HID) / 256, 256, 0, stream>>>(agg, b3, h);

  // ---- layer 4 (Nout padded to one 16-wide tile; single wave per block)
  k_gemm_wmma_lds<<<MT, 32, 0, stream>>>(h, wt4, sup4, 16);
  k_zero<<<(N_NODES * 8) / 256, 256, 0, stream>>>(agg4, N_NODES * 8);
  k_spmm6<<<(N_EDGES * 8) / 256, 256, 0, stream>>>(erow, ecol, ew, sup4, agg4);
  k_final<<<(N_NODES * N_CLASS + 255) / 256, 256, 0, stream>>>(agg4, b4, out);
}
